// GRASSLayer_88570815578174
// MI455X (gfx1250) — matrix operand order repaired and checked by reading, not compiled, not run on hardware
//
#include <hip/hip_runtime.h>
#include <hip/hip_bf16.h>

// ---------------------------------------------------------------------------
// GRASS GNN layer for MI455X (gfx1250, wave32, WMMA).
// All GEMMs run through V_WMMA_F32_16X16X32_BF16 (bf16 inputs, f32 accum).
// Reductions / softmax / mish / residuals stay in f32.
// Round 2: per-block row-tile loops (W staged once per block), segment-max
// fused into the edge input GEMM, global_prefetch of the next A tile.
// ---------------------------------------------------------------------------

typedef __attribute__((ext_vector_type(16))) __bf16   v16bf;
typedef __attribute__((ext_vector_type(8)))  float    v8f;
typedef __attribute__((ext_vector_type(8)))  unsigned v8u;

#define K_EPS      1e-7f
#define K_ATTN_EPS 3.1622776601683794e-4f  // sqrt(1e-7)

__device__ __forceinline__ unsigned bf16rne(float f) {
  unsigned u = __float_as_uint(f);
  return (u + 0x7fffu + ((u >> 16) & 1u)) >> 16;   // round-to-nearest-even
}
__device__ __forceinline__ unsigned packbf(float a, float b) {
  return bf16rne(a) | (bf16rne(b) << 16);
}
__device__ __forceinline__ float mishf(float v) {
  float sp = (v > 20.f) ? v : log1pf(__expf(v));   // softplus
  return v * tanhf(sp);
}
// Monotone float<->uint key so atomicMax(u32) implements segment_max exactly
// (and stays bitwise deterministic regardless of arrival order).
__device__ __forceinline__ unsigned f2key(float f) {
  unsigned u = __float_as_uint(f);
  return (u & 0x80000000u) ? ~u : (u | 0x80000000u);
}
__device__ __forceinline__ float key2f(unsigned k) {
  unsigned u = (k & 0x80000000u) ? (k & 0x7fffffffu) : ~k;
  return __uint_as_float(u);
}

// ---------------------------------------------------------------------------
// Shared WMMA core: one 128-thread block = 4 waves; wave w computes the
// 16-row x 128-col tile rows [16w,16w+16).  A: [64 rows][K=128] bf16 in LDS,
// Wt: [128 cols][K=128] bf16 in LDS (column-major W so B K-pairs are
// contiguous).  Pitch 130 (260B) -> LDS bank stride 65 == 1 mod 64.
// ISA layouts (CDNA5 ISA 7.12.2):
//   A v[j], lane L: row L%16, K-pair index ks/2 + 4*(L/16) + (j<4 ? j : 8+j-4)
//   B v[j], lane L: col L%16, K-pair index ks/2 + 4*(L/16) + (j<4 ? j : 8+j-4)
//   C v[j], lane L: row j + 8*(L/16),     col L%16
// ---------------------------------------------------------------------------
__device__ __forceinline__ void wmma_tile_128(const unsigned short (*A)[130],
                                              const unsigned short (*Wt)[130],
                                              v8f acc[8]) {
  const int lane = threadIdx.x & 31;
  const int lr   = lane & 15;
  const int hf   = lane >> 4;
  const int r0   = ((threadIdx.x >> 5) & 3) << 4;   // wave id * 16
#pragma unroll
  for (int kk = 0; kk < 4; ++kk) {
    const int jb = kk * 16;                         // K-pair base (= ks/2)
    const unsigned short* arow = A[r0 + lr];
    const int ab = jb + hf * 4;
    v8u au;
#pragma unroll
    for (int j = 0; j < 4; ++j) {
      au[j]     = *(const unsigned*)(arow + 2 * (ab + j));
      au[j + 4] = *(const unsigned*)(arow + 2 * (ab + 8 + j));
    }
    const v16bf av = __builtin_bit_cast(v16bf, au);
    const int bb = jb + hf * 4;
#pragma unroll
    for (int n = 0; n < 8; ++n) {
      const unsigned short* brow = Wt[16 * n + lr];
      v8u bu;
#pragma unroll
      for (int j = 0; j < 8; ++j)
        bu[j] = *(const unsigned*)(brow + 2 * (bb + j + (j & 4)));
      const v16bf bv = __builtin_bit_cast(v16bf, bu);
      acc[n] = __builtin_amdgcn_wmma_f32_16x16x32_bf16(
          false, av, false, bv, (short)0, acc[n], false, false);
    }
  }
}

// Stage a 128(K) x 128(col) slice of W (row-major, leading dim ldW) into LDS,
// transposed + bf16-packed: Wt[col][k-pair].  128 threads <-> 128 columns;
// each global load is lane-coalesced (consecutive t -> consecutive cols).
__device__ __forceinline__ void stage_w(const float* __restrict__ W, int ldW,
                                        int cb, unsigned short (*Wt)[130]) {
  const int t = threadIdx.x;
#pragma unroll 4
  for (int k2 = 0; k2 < 64; ++k2) {
    const float a = W[(size_t)(2 * k2) * ldW + cb + t];
    const float b = W[(size_t)(2 * k2 + 1) * ldW + cb + t];
    *(unsigned*)&Wt[t][2 * k2] = packbf(a, b);
  }
}

// ---------------------------------------------------------------------------
// Kernel 1/2: Y[M x ldY] = X[M x 128] @ W[128 x ldY]   (f32 in, f32 out)
// grid = (min(ntiles,CAP), ldY/128), block = 128; each block loops over row
// tiles with stride gridDim.x so Wt is staged once per block.
// If segkey != nullptr, the col group `attn_group` additionally feeds the
// per-(edge,channel) atomic segment-max keyed by tail[] (fused seg_max).
// ---------------------------------------------------------------------------
__global__ void __launch_bounds__(128)
gemm_in_kernel(const float* __restrict__ X, const float* __restrict__ W,
               float* __restrict__ Y, int M, int ldY,
               const int* __restrict__ tail, unsigned* __restrict__ segkey,
               int attn_group) {
  __shared__ unsigned short Wt[128][130];
  __shared__ unsigned short A[64][130];
  const int cb = blockIdx.y * 128;
  stage_w(W, ldY, cb, Wt);

  const int ntiles = (M + 63) >> 6;
  const bool do_attn = (segkey != nullptr) && (blockIdx.y == (unsigned)attn_group);

  const int lane = threadIdx.x & 31, lr = lane & 15, hf = lane >> 4;
  const int r0 = ((threadIdx.x >> 5) & 3) << 4;

  for (int tile = blockIdx.x; tile < ntiles; tile += gridDim.x) {
    const int row0 = tile << 6;
    __syncthreads();                                  // A reuse from prev iter
    // Prefetch next tile's X rows (gfx1250 global_prefetch_b8).
    const int nrow0 = row0 + ((int)gridDim.x << 6);
    if (nrow0 + 64 <= M) {
      const char* pb = (const char*)(X + (size_t)nrow0 * 128);
      __builtin_prefetch(pb + threadIdx.x * 128, 0, 0);
      __builtin_prefetch(pb + 16384 + threadIdx.x * 128, 0, 0);
    }
    for (int i = threadIdx.x; i < 64 * 64; i += 128) { // 64 rows x 64 pairs
      const int r = i >> 6, jp = i & 63;
      int row = row0 + r;
      if (row >= M) row = M - 1;                       // clamp (store guarded)
      const float2 xv = *(const float2*)(X + (size_t)row * 128 + 2 * jp);
      *(unsigned*)&A[r][2 * jp] = packbf(xv.x, xv.y);
    }
    __syncthreads();

    v8f acc[8] = {};
    wmma_tile_128(A, Wt, acc);

#pragma unroll
    for (int j = 0; j < 8; ++j) {
      const int grow = row0 + r0 + j + 8 * hf;
      if (grow < M) {
        const int tl = do_attn ? tail[grow] : 0;
#pragma unroll
        for (int n = 0; n < 8; ++n) {
          const int col = cb + 16 * n + lr;
          const float v = acc[n][j];
          Y[(size_t)grow * ldY + col] = v;
          if (do_attn)
            atomicMax(segkey + (size_t)tl * 128 + (col - cb), f2key(v));
        }
      }
    }
  }
}

// ---------------------------------------------------------------------------
// Kernel 0: zero segmax keys + received + recv_weight (key 0 == -NaN sentinel,
// never read back because every read site has >=1 contributing edge).
// ---------------------------------------------------------------------------
__global__ void zero_kernel(unsigned* __restrict__ p, long long n) {
  const long long i = (long long)blockIdx.x * 256 + threadIdx.x;
  if (i < n) p[i] = 0u;
}

// Kernel 3: attention weight + scatter-add of sent / recv_weight.
// 256-thread block handles 2 edges (threadIdx.x/128 selects the edge).
__global__ void __launch_bounds__(256)
scatter_kernel(const float* __restrict__ edge_msgs,
               const float* __restrict__ node_msgs,
               const int* __restrict__ head, const int* __restrict__ tail,
               const unsigned* __restrict__ segkey,
               float* __restrict__ received, float* __restrict__ recvw, int E) {
  const int e = blockIdx.x * 2 + (threadIdx.x >> 7);
  if (e >= E) return;
  const int c = threadIdx.x & 127;
  const int t = tail[e], h = head[e];
  const float att = edge_msgs[(size_t)e * 384 + 256 + c];
  const float m   = key2f(segkey[(size_t)t * 128 + c]);
  const float w   = expf(att - m) + K_ATTN_EPS;
  const float s   = (node_msgs[(size_t)h * 512 + 128 + c] +
                     edge_msgs[(size_t)e * 384 + 128 + c]) * w;
  atomicAdd(received + (size_t)t * 128 + c, s);
  atomicAdd(recvw    + (size_t)t * 128 + c, w);
}

// ---------------------------------------------------------------------------
// Kernel 4: node output path.  pre = received/(w+eps) + self + bias;
// out = x + mish(pre) @ W_node_out + b.   Row-tile loop per block.
// ---------------------------------------------------------------------------
__global__ void __launch_bounds__(128)
node_out_kernel(const float* __restrict__ x, const float* __restrict__ node_msgs,
                const float* __restrict__ received, const float* __restrict__ recvw,
                const float* __restrict__ nbias, const float* __restrict__ Wout,
                const float* __restrict__ bout, float* __restrict__ out, int M) {
  __shared__ unsigned short Wt[128][130];
  __shared__ unsigned short A[64][130];
  stage_w(Wout, 128, 0, Wt);

  const int ntiles = (M + 63) >> 6;
  const int lane = threadIdx.x & 31, lr = lane & 15, hf = lane >> 4;
  const int r0 = ((threadIdx.x >> 5) & 3) << 4;

  for (int tile = blockIdx.x; tile < ntiles; tile += gridDim.x) {
    const int row0 = tile << 6;
    __syncthreads();
    for (int i = threadIdx.x; i < 64 * 128; i += 128) {
      const int r = i >> 7, c = i & 127;
      int row = row0 + r;
      if (row >= M) row = M - 1;
      const float pre = received[(size_t)row * 128 + c] /
                            (recvw[(size_t)row * 128 + c] + K_EPS) +
                        node_msgs[(size_t)row * 512 + c] + nbias[c];
      A[r][c] = (unsigned short)bf16rne(mishf(pre));
    }
    __syncthreads();

    v8f acc[8] = {};
    wmma_tile_128(A, Wt, acc);

#pragma unroll
    for (int n = 0; n < 8; ++n) {
      const int col = 16 * n + lr;
#pragma unroll
      for (int j = 0; j < 8; ++j) {
        const int grow = row0 + r0 + j + 8 * hf;
        if (grow < M)
          out[(size_t)grow * 128 + col] =
              x[(size_t)grow * 128 + col] + acc[n][j] + bout[col];
      }
    }
  }
}

// ---------------------------------------------------------------------------
// Kernel 5: edge output path.  pre = head_to_edge[head] + tail_to_edge[tail]
// + edge_to_self + bias; out = edge_attr + mish(pre) @ W_edge_out + b.
// ---------------------------------------------------------------------------
__global__ void __launch_bounds__(128)
edge_out_kernel(const float* __restrict__ edge_attr,
                const float* __restrict__ node_msgs,
                const float* __restrict__ edge_msgs,
                const int* __restrict__ head, const int* __restrict__ tail,
                const float* __restrict__ ebias, const float* __restrict__ Wout,
                const float* __restrict__ bout, float* __restrict__ out, int M) {
  __shared__ unsigned short Wt[128][130];
  __shared__ unsigned short A[64][130];
  stage_w(Wout, 128, 0, Wt);

  const int ntiles = (M + 63) >> 6;
  const int lane = threadIdx.x & 31, lr = lane & 15, hf = lane >> 4;
  const int r0 = ((threadIdx.x >> 5) & 3) << 4;

  for (int tile = blockIdx.x; tile < ntiles; tile += gridDim.x) {
    const int row0 = tile << 6;
    __syncthreads();
    for (int i = threadIdx.x; i < 64 * 128; i += 128) {
      const int r = i >> 7, c = i & 127;
      int row = row0 + r;
      if (row >= M) row = M - 1;
      const int h = head[row], t = tail[row];
      const float pre = node_msgs[(size_t)h * 512 + 256 + c] +
                        node_msgs[(size_t)t * 512 + 384 + c] +
                        edge_msgs[(size_t)row * 384 + c] + ebias[c];
      A[r][c] = (unsigned short)bf16rne(mishf(pre));
    }
    __syncthreads();

    v8f acc[8] = {};
    wmma_tile_128(A, Wt, acc);

#pragma unroll
    for (int n = 0; n < 8; ++n) {
      const int col = 16 * n + lr;
#pragma unroll
      for (int j = 0; j < 8; ++j) {
        const int grow = row0 + r0 + j + 8 * hf;
        if (grow < M)
          out[(size_t)grow * 128 + col] =
              edge_attr[(size_t)grow * 128 + col] + acc[n][j] + bout[col];
      }
    }
  }
}

// ---------------------------------------------------------------------------
extern "C" void kernel_launch(void* const* d_in, const int* in_sizes, int n_in,
                              void* d_out, int out_size, void* d_ws, size_t ws_size,
                              hipStream_t stream) {
  (void)n_in; (void)out_size; (void)ws_size;
  const float* x          = (const float*)d_in[0];
  const float* edge_attr  = (const float*)d_in[1];
  const int*   head       = (const int*)d_in[2];
  const int*   tail       = (const int*)d_in[3];
  // d_in[4] node_batch, d_in[5] edge_batch, d_in[6] num_nodes, d_in[7]
  // num_graphs: unused by the forward computation.
  const float* W_node_in  = (const float*)d_in[8];
  const float* W_edge_in  = (const float*)d_in[9];
  const float* nbias      = (const float*)d_in[10];
  const float* ebias      = (const float*)d_in[11];
  const float* W_node_out = (const float*)d_in[12];
  const float* b_node_out = (const float*)d_in[13];
  const float* W_edge_out = (const float*)d_in[14];
  const float* b_edge_out = (const float*)d_in[15];

  const int N = in_sizes[0] / 128;
  const int E = in_sizes[1] / 128;

  // Workspace layout (~1.13 GB):
  float*    node_msgs = (float*)d_ws;                             // N x 512
  float*    edge_msgs = node_msgs + (size_t)N * 512;              // E x 384
  unsigned* segkey    = (unsigned*)(edge_msgs + (size_t)E * 384); // N x 128
  float*    received  = (float*)(segkey + (size_t)N * 128);       // N x 128
  float*    recvw     = received + (size_t)N * 128;               // N x 128

  float* x_out = (float*)d_out;                // outputs concatenated flat
  float* e_out = x_out + (size_t)N * 128;

  const long long zn = (long long)N * 128 * 3; // segkey + received + recvw
  zero_kernel<<<(unsigned)((zn + 255) / 256), 256, 0, stream>>>(segkey, zn);

  const int ntN = (N + 63) / 64, ntE = (E + 63) / 64;
  dim3 g1((unsigned)(ntN < 768 ? ntN : 768), 4);
  gemm_in_kernel<<<g1, 128, 0, stream>>>(x, W_node_in, node_msgs, N, 512,
                                         nullptr, nullptr, -1);
  // Edge input GEMM with fused segment-max on col group 2 (attn chunk).
  dim3 g2((unsigned)(ntE < 768 ? ntE : 768), 3);
  gemm_in_kernel<<<g2, 128, 0, stream>>>(edge_attr, W_edge_in, edge_msgs, E, 384,
                                         tail, segkey, 2);

  scatter_kernel<<<(E + 1) / 2, 256, 0, stream>>>(edge_msgs, node_msgs, head,
                                                  tail, segkey, received, recvw, E);

  node_out_kernel<<<(ntN < 1024 ? ntN : 1024), 128, 0, stream>>>(
      x, node_msgs, received, recvw, nbias, W_node_out, b_node_out, x_out, N);
  edge_out_kernel<<<(ntE < 1024 ? ntE : 1024), 128, 0, stream>>>(
      edge_attr, node_msgs, edge_msgs, head, tail, ebias, W_edge_out,
      b_edge_out, e_out, E);
}
